// SAINetV2_43490838839385
// MI455X (gfx1250) — matrix operand order repaired and checked by reading, not compiled, not run on hardware
//
#include <hip/hip_runtime.h>
#include <hip/hip_bf16.h>
#include <math.h>

// ---------------------------------------------------------------------------
// Problem constants (match reference)
// ---------------------------------------------------------------------------
#define NN   30000
#define EE   120000
#define GG   1200
#define HIDD 256
#define LLAY 6
#define NHEAD 4
#define NTASK 12

#define NP   30080    // NN padded to 128
#define EPD  120064   // EE padded to 128
#define GP   1280     // GG padded to 128

typedef __bf16 bf16;
typedef __attribute__((ext_vector_type(16))) __bf16 v16bf;
typedef __attribute__((ext_vector_type(8)))  __bf16 v8bf;
typedef __attribute__((ext_vector_type(8)))  float  v8f;
typedef int v4i __attribute__((vector_size(16)));

// ---------------------------------------------------------------------------
// CDNA5 async-copy helpers (guarded; fall back to sync copy if absent)
// ---------------------------------------------------------------------------
__device__ __forceinline__ void async_copy16(const bf16* g, bf16* l) {
#if defined(__has_builtin) && __has_builtin(__builtin_amdgcn_global_load_async_to_lds_b128)
  __builtin_amdgcn_global_load_async_to_lds_b128(
      (__attribute__((address_space(1))) v4i*)g,
      (__attribute__((address_space(3))) v4i*)l, 0, 0);
#else
  *(uint4*)l = *(const uint4*)g;
#endif
}

__device__ __forceinline__ void wait_async0() {
#if defined(__has_builtin) && __has_builtin(__builtin_amdgcn_s_wait_asynccnt)
  __builtin_amdgcn_s_wait_asynccnt(0);
#else
  asm volatile("s_wait_asynccnt 0" ::: "memory");
#endif
}

__device__ __forceinline__ float silu_f(float x) { return x / (1.0f + expf(-x)); }

// ordered-uint encoding for float atomicMax
__device__ __forceinline__ unsigned f2ord(float f) {
  unsigned u = __float_as_uint(f);
  return (u & 0x80000000u) ? ~u : (u | 0x80000000u);
}
__device__ __forceinline__ float ord2f(unsigned k) {
  unsigned u = (k & 0x80000000u) ? (k & 0x7FFFFFFFu) : ~k;
  return __uint_as_float(u);
}

// ---------------------------------------------------------------------------
// WMMA bf16 GEMM:  C[M,N] = act(A[M,K] @ W[K,N] + bias)
// A, W are bf16; C is f32.  M must be a multiple of 128 (padded buffers),
// N a multiple of 64, K a multiple of 32 (K <= 512).
// Block = 256 threads (8 waves). Tile: 128(M) x 64(N). Each wave owns 16 rows
// and all 64 cols (4 wmma accumulators).
// ---------------------------------------------------------------------------
#define BM 128
#define BN 64
#define BK 32
#define KMAX 512

__device__ __forceinline__ void stage_tileA(const bf16* __restrict__ A, bf16* At,
                                            int tid, int m0, int k0, int K) {
  // 128 rows x 32 bf16 = 512 x 16B chunks; 256 threads -> 2 each
#pragma unroll
  for (int it = 0; it < 2; ++it) {
    int i   = tid + it * 256;
    int row = i >> 2;
    int cc  = i & 3;
    const bf16* g = A + (size_t)(m0 + row) * K + k0 + cc * 8;
    bf16*       l = At + row * BK + cc * 8;
    async_copy16(g, l);
  }
}

template <int ACT>
__global__ __launch_bounds__(256) void gemm_bf16_wmma(
    const bf16* __restrict__ A,     // [Mpad, K]
    const bf16* __restrict__ W,     // [K, N] row-major
    const float* __restrict__ bias, // [N] or nullptr
    float* __restrict__ C,          // [Mpad, N]
    int K, int N) {
  __shared__ bf16 Wp[BN * KMAX];      // transposed panel: Wp[n*K + k]
  __shared__ bf16 At[2][BM * BK];

  const int tid  = threadIdx.x;
  const int m0   = blockIdx.x * BM;
  const int n0   = blockIdx.y * BN;
  const int wave = tid >> 5;
  const int lane = tid & 31;
  const int lhalf = lane >> 4;  // 0: lanes 0-15, 1: lanes 16-31
  const int l16   = lane & 15;

  // Stage W panel, transposed ([n][k]) so B fragments are contiguous in LDS.
  for (int idx = tid; idx < BN * K; idx += 256) {
    int k = idx >> 6;   // / 64
    int n = idx & 63;
    Wp[n * K + k] = W[(size_t)k * N + (n0 + n)];
  }

  stage_tileA(A, At[0], tid, m0, 0, K);
  wait_async0();
  __syncthreads();

  v8f acc[4] = {};

  const int nk = K >> 5;
  for (int t = 0; t < nk; ++t) {
    const int buf = t & 1;
    if (t + 1 < nk) stage_tileA(A, At[buf ^ 1], tid, m0, (t + 1) << 5, K);

    const int k0 = t << 5;
    // A fragment (16x32 bf16): lane m = l16; K lhalf*16 .. +15 (contiguous, matches B).
    const bf16* arow = &At[buf][(wave * 16 + l16) * BK];
    v8bf alo = *(const v8bf*)(arow + (lhalf ? 16 : 0));
    v8bf ahi = *(const v8bf*)(arow + (lhalf ? 24 : 8));
    v16bf afrag = __builtin_shufflevector(alo, ahi, 0, 1, 2, 3, 4, 5, 6, 7,
                                          8, 9, 10, 11, 12, 13, 14, 15);

    // Preload all four B fragments first so a single dscnt wait covers the
    // whole WMMA chain (back-to-back XDL issue instead of load-wait-wmma).
    v16bf bfr[4];
#pragma unroll
    for (int nb = 0; nb < 4; ++nb) {
      // B fragment (32x16): lane owns column nb*16+l16, K k0+lhalf*16 .. +15
      bfr[nb] = *(const v16bf*)&Wp[(nb * 16 + l16) * K + k0 + lhalf * 16];
    }
#pragma unroll
    for (int nb = 0; nb < 4; ++nb) {
      acc[nb] = __builtin_amdgcn_wmma_f32_16x16x32_bf16(
          false, afrag, false, bfr[nb], (short)0, acc[nb], false, false);
    }

    if (t + 1 < nk) wait_async0();
    __syncthreads();
  }

  // Epilogue per ISA C/D layout: acc[nb] VGPR r -> row (lhalf?8+r:r), col l16.
#pragma unroll
  for (int nb = 0; nb < 4; ++nb) {
    const int col = n0 + nb * 16 + l16;
    const float bb = bias ? bias[col] : 0.0f;
#pragma unroll
    for (int r = 0; r < 8; ++r) {
      const int row = m0 + wave * 16 + lhalf * 8 + r;
      float v = acc[nb][r] + bb;
      if (ACT == 1) v = silu_f(v);
      C[(size_t)row * N + col] = v;
    }
  }
}

// ---------------------------------------------------------------------------
// f32 -> bf16 conversion (one element per thread)
// ---------------------------------------------------------------------------
__global__ void cvt_bf16_kernel(const float* __restrict__ in, bf16* __restrict__ out,
                                size_t n) {
  size_t i = (size_t)blockIdx.x * blockDim.x + threadIdx.x;
  if (i < n) out[i] = (bf16)in[i];
}

// ---------------------------------------------------------------------------
// Small-K embedding: out = X[M,K] @ W[K,256] + b ; writes f32 and/or bf16
// Block = 256 (one output row per block).
// ---------------------------------------------------------------------------
__global__ void embed_kernel(const float* __restrict__ X, const float* __restrict__ W,
                             const float* __restrict__ b, float* __restrict__ outF,
                             bf16* __restrict__ outB, int K) {
  const int row = blockIdx.x;
  const int j = threadIdx.x;
  __shared__ float xr[64];
  if (j < K) xr[j] = X[(size_t)row * K + j];
  __syncthreads();
  float acc = b ? b[j] : 0.0f;
  for (int k = 0; k < K; ++k) acc += xr[k] * W[k * HIDD + j];
  if (outF) outF[(size_t)row * HIDD + j] = acc;
  if (outB) outB[(size_t)row * HIDD + j] = (bf16)acc;
}

// ---------------------------------------------------------------------------
// Per-layer init: agg <- b_conv (broadcast), amax <- ord(-inf), denom <- 0
// ---------------------------------------------------------------------------
__global__ void init_layer_kernel(float* __restrict__ agg, unsigned* __restrict__ amax,
                                  float* __restrict__ denom,
                                  const float* __restrict__ bconv, int n, int n4) {
  int i = blockIdx.x * blockDim.x + threadIdx.x;
  if (i < n) agg[i] = bconv[i & (HIDD - 1)];
  if (i < n4) { amax[i] = f2ord(-3.0e38f); denom[i] = 0.0f; }
}

// ---------------------------------------------------------------------------
// GATv2 logits: m = leaky_relu(xl[src]+xr[dst]+e, 0.2); logit = <m, att_h>
// Block = 256 threads, one edge per block. Also atomic-max into amax[dst][h].
// ---------------------------------------------------------------------------
__global__ void attn_logits_kernel(const float* __restrict__ xl,
                                   const float* __restrict__ xr,
                                   const float* __restrict__ ep,
                                   const int* __restrict__ src,
                                   const int* __restrict__ dst,
                                   const float* __restrict__ att,  // [256]
                                   float* __restrict__ logits,     // [E,4]
                                   unsigned* __restrict__ amax) {  // [N,4]
  const int e = blockIdx.x;
  const int c = threadIdx.x;
  const int s = src[e], d = dst[e];
  float m = xl[(size_t)s * HIDD + c] + xr[(size_t)d * HIDD + c] +
            ep[(size_t)e * HIDD + c];
  m = (m > 0.0f) ? m : 0.2f * m;
  float p = m * att[c];
  __shared__ float red[HIDD];
  red[c] = p;
  for (int off = 32; off > 0; off >>= 1) {
    __syncthreads();
    if ((c & 63) < off) red[c] += red[c + off];
  }
  __syncthreads();
  if ((c & 63) == 0) {
    const int hh = c >> 6;
    const float lg = red[c];
    logits[(size_t)e * NHEAD + hh] = lg;
    atomicMax(&amax[(size_t)d * NHEAD + hh], f2ord(lg));
  }
}

// ---------------------------------------------------------------------------
// w = exp(logit - amax[dst]); denom[dst] += w
// ---------------------------------------------------------------------------
__global__ void attn_exp_kernel(const float* __restrict__ logits,
                                const unsigned* __restrict__ amax,
                                const int* __restrict__ dst,
                                float* __restrict__ wbuf,
                                float* __restrict__ denom, int total) {
  int i = blockIdx.x * blockDim.x + threadIdx.x;
  if (i >= total) return;
  const int e = i >> 2;
  const int h = i & 3;
  const int d = dst[e];
  float w = expf(logits[i] - ord2f(amax[(size_t)d * NHEAD + h]));
  wbuf[i] = w;
  atomicAdd(&denom[(size_t)d * NHEAD + h], w);
}

// ---------------------------------------------------------------------------
// agg[dst] += xl[src] * alpha   (alpha = w / (denom + 1e-16))
// ---------------------------------------------------------------------------
__global__ void aggregate_kernel(const float* __restrict__ xl,
                                 const float* __restrict__ wbuf,
                                 const float* __restrict__ denom,
                                 const int* __restrict__ src,
                                 const int* __restrict__ dst,
                                 float* __restrict__ agg) {
  const int e = blockIdx.x;
  const int c = threadIdx.x;
  const int s = src[e], d = dst[e];
  const int h = c >> 6;
  float alpha = wbuf[(size_t)e * NHEAD + h] /
                (denom[(size_t)d * NHEAD + h] + 1e-16f);
  atomicAdd(&agg[(size_t)d * HIDD + c], xl[(size_t)s * HIDD + c] * alpha);
}

// ---------------------------------------------------------------------------
// h <- layer_norm(h + agg) * g + b   (one node per 256-thread block)
// ---------------------------------------------------------------------------
__global__ void ln_kernel(float* __restrict__ h, const float* __restrict__ agg,
                          const float* __restrict__ g, const float* __restrict__ b) {
  const int n = blockIdx.x;
  const int c = threadIdx.x;
  float v = h[(size_t)n * HIDD + c] + agg[(size_t)n * HIDD + c];
  __shared__ float s1[HIDD], s2[HIDD];
  s1[c] = v;
  s2[c] = v * v;
  for (int off = 128; off > 0; off >>= 1) {
    __syncthreads();
    if (c < off) { s1[c] += s1[c + off]; s2[c] += s2[c + off]; }
  }
  __syncthreads();
  const float mu  = s1[0] * (1.0f / HIDD);
  const float var = s2[0] * (1.0f / HIDD) - mu * mu;
  const float r   = rsqrtf(var + 1e-5f);
  h[(size_t)n * HIDD + c] = (v - mu) * r * g[c] + b[c];
}

// ---------------------------------------------------------------------------
// Pooling
// ---------------------------------------------------------------------------
__global__ void pool_init_kernel(float* __restrict__ ssum, float* __restrict__ cnt,
                                 int n, int ng) {
  int i = blockIdx.x * blockDim.x + threadIdx.x;
  if (i < n) ssum[i] = 0.0f;
  if (i < ng) cnt[i] = 0.0f;
}

__global__ void pool_acc_kernel(const float* __restrict__ h,
                                const int* __restrict__ batch,
                                float* __restrict__ ssum, float* __restrict__ cnt) {
  const int n = blockIdx.x;
  const int c = threadIdx.x;
  const int b = batch[n];
  atomicAdd(&ssum[(size_t)b * HIDD + c], h[(size_t)n * HIDD + c]);
  if (c == 0) atomicAdd(&cnt[b], 1.0f);
}

__global__ void pool_fin_kernel(const float* __restrict__ ssum,
                                const float* __restrict__ cnt,
                                bf16* __restrict__ gcat) {
  const int g = blockIdx.x;
  const int c = threadIdx.x;
  const float s = ssum[(size_t)g * HIDD + c];
  const float mean = s / fmaxf(cnt[g], 1.0f);
  gcat[(size_t)g * 512 + c] = (bf16)mean;
  gcat[(size_t)g * 512 + HIDD + c] = (bf16)s;
}

// ---------------------------------------------------------------------------
// Task heads
// ---------------------------------------------------------------------------
__global__ void head1_kernel(const float* __restrict__ z2, const float* __restrict__ W,
                             const float* __restrict__ b, float* __restrict__ h1) {
  const int g = blockIdx.x, t = blockIdx.y, k = threadIdx.x;  // k < 128
  const float* zr = z2 + (size_t)g * HIDD;
  const float* Wt = W + (size_t)t * HIDD * 128;
  float acc = b[t * 128 + k];
  for (int d = 0; d < HIDD; ++d) acc += zr[d] * Wt[d * 128 + k];
  h1[((size_t)g * NTASK + t) * 128 + k] = silu_f(acc);
}

__global__ void head2_kernel(const float* __restrict__ h1, const float* __restrict__ W,
                             const float* __restrict__ b, float* __restrict__ h2) {
  const int g = blockIdx.x, t = blockIdx.y, m = threadIdx.x;  // m < 64
  const float* hr = h1 + ((size_t)g * NTASK + t) * 128;
  const float* Wt = W + (size_t)t * 128 * 64;
  float acc = b[t * 64 + m];
  for (int k = 0; k < 128; ++k) acc += hr[k] * Wt[k * 64 + m];
  h2[((size_t)g * NTASK + t) * 64 + m] = silu_f(acc);
}

__global__ void head3_kernel(const float* __restrict__ h2, const float* __restrict__ W3,
                             const float* __restrict__ b3, float* __restrict__ out,
                             int total) {
  int i = blockIdx.x * blockDim.x + threadIdx.x;
  if (i >= total) return;
  const int t = i % NTASK;
  const float* hr = h2 + (size_t)i * 64;
  const float* w = W3 + t * 64;
  float acc = b3[t];
  for (int m = 0; m < 64; ++m) acc += hr[m] * w[m];
  out[i] = acc;
}

// ---------------------------------------------------------------------------
// Host launcher
// ---------------------------------------------------------------------------
static inline size_t align256(size_t x) { return (x + 255) & ~(size_t)255; }

extern "C" void kernel_launch(void* const* d_in, const int* in_sizes, int n_in,
                              void* d_out, int out_size, void* d_ws, size_t ws_size,
                              hipStream_t stream) {
  (void)in_sizes; (void)n_in; (void)out_size; (void)ws_size;

  // Inputs (setup_inputs order)
  const float* x          = (const float*)d_in[0];
  const int*   edge_index = (const int*)d_in[1];
  const float* edge_attr  = (const float*)d_in[2];
  const int*   batch      = (const int*)d_in[3];
  const float* W_node     = (const float*)d_in[4];
  const float* b_node     = (const float*)d_in[5];
  const float* W_edge     = (const float*)d_in[6];
  const float* b_edge     = (const float*)d_in[7];
  const float* Wl         = (const float*)d_in[8];
  const float* bl         = (const float*)d_in[9];
  const float* Wr         = (const float*)d_in[10];
  const float* br         = (const float*)d_in[11];
  const float* We         = (const float*)d_in[12];
  const float* att        = (const float*)d_in[13];
  const float* b_conv     = (const float*)d_in[14];
  const float* ln_g       = (const float*)d_in[15];
  const float* ln_b       = (const float*)d_in[16];
  const float* W_fc1      = (const float*)d_in[17];
  const float* b_fc1      = (const float*)d_in[18];
  const float* W_fc2      = (const float*)d_in[19];
  const float* b_fc2      = (const float*)d_in[20];
  const float* W_h1       = (const float*)d_in[21];
  const float* b_h1       = (const float*)d_in[22];
  const float* W_h2       = (const float*)d_in[23];
  const float* b_h2       = (const float*)d_in[24];
  const float* W_h3       = (const float*)d_in[25];
  const float* b_h3       = (const float*)d_in[26];

  const int* src = edge_index;
  const int* dst = edge_index + EE;
  float* out = (float*)d_out;

  // Workspace carve
  char* p = (char*)d_ws;
  auto carve = [&](size_t bytes) -> void* { void* r = (void*)p; p += align256(bytes); return r; };

  float*    h      = (float*)carve((size_t)NP * HIDD * 4);
  bf16*     h_bf   = (bf16*)carve((size_t)NP * HIDD * 2);
  bf16*     e0_bf  = (bf16*)carve((size_t)EPD * HIDD * 2);
  float*    xl     = (float*)carve((size_t)NP * HIDD * 4);
  float*    xr     = (float*)carve((size_t)NP * HIDD * 4);
  float*    ep     = (float*)carve((size_t)EPD * HIDD * 4);
  float*    logits = (float*)carve((size_t)EE * NHEAD * 4);
  float*    wbuf   = (float*)carve((size_t)EE * NHEAD * 4);
  unsigned* amax   = (unsigned*)carve((size_t)NN * NHEAD * 4);
  float*    denom  = (float*)carve((size_t)NN * NHEAD * 4);
  float*    agg    = (float*)carve((size_t)NP * HIDD * 4);
  bf16*     Wl_bf  = (bf16*)carve((size_t)LLAY * HIDD * HIDD * 2);
  bf16*     Wr_bf  = (bf16*)carve((size_t)LLAY * HIDD * HIDD * 2);
  bf16*     We_bf  = (bf16*)carve((size_t)LLAY * HIDD * HIDD * 2);
  bf16*     Wf1_bf = (bf16*)carve((size_t)512 * HIDD * 2);
  bf16*     Wf2_bf = (bf16*)carve((size_t)HIDD * HIDD * 2);
  float*    ssum   = (float*)carve((size_t)GG * HIDD * 4);
  float*    cnt    = (float*)carve((size_t)GG * 4);
  bf16*     gcat   = (bf16*)carve((size_t)GP * 512 * 2);
  float*    z1     = (float*)carve((size_t)GP * HIDD * 4);
  bf16*     z1_bf  = (bf16*)carve((size_t)GP * HIDD * 2);
  float*    z2     = (float*)carve((size_t)GP * HIDD * 4);
  float*    h1     = (float*)carve((size_t)GG * NTASK * 128 * 4);
  float*    h2     = (float*)carve((size_t)GG * NTASK * 64 * 4);

  auto cvt = [&](const float* in, bf16* o, size_t n) {
    cvt_bf16_kernel<<<dim3((unsigned)((n + 255) / 256)), dim3(256), 0, stream>>>(in, o, n);
  };

  // Convert all GEMM weights to bf16 once per call
  cvt(Wl, Wl_bf, (size_t)LLAY * HIDD * HIDD);
  cvt(Wr, Wr_bf, (size_t)LLAY * HIDD * HIDD);
  cvt(We, We_bf, (size_t)LLAY * HIDD * HIDD);
  cvt(W_fc1, Wf1_bf, (size_t)512 * HIDD);
  cvt(W_fc2, Wf2_bf, (size_t)HIDD * HIDD);

  // Embeddings
  embed_kernel<<<dim3(NN), dim3(256), 0, stream>>>(x, W_node, b_node, h, nullptr, 41);
  embed_kernel<<<dim3(EE), dim3(256), 0, stream>>>(edge_attr, W_edge, b_edge, nullptr, e0_bf, 10);

  const dim3 gemm_grid_n(NP / BM, HIDD / BN);   // (235, 4)
  const dim3 gemm_grid_e(EPD / BM, HIDD / BN);  // (938, 4)
  const dim3 gemm_grid_g(GP / BM, HIDD / BN);   // (10, 4)

  for (int i = 0; i < LLAY; ++i) {
    cvt(h, h_bf, (size_t)NN * HIDD);

    gemm_bf16_wmma<0><<<gemm_grid_n, dim3(256), 0, stream>>>(
        h_bf, Wl_bf + (size_t)i * HIDD * HIDD, bl + i * HIDD, xl, HIDD, HIDD);
    gemm_bf16_wmma<0><<<gemm_grid_n, dim3(256), 0, stream>>>(
        h_bf, Wr_bf + (size_t)i * HIDD * HIDD, br + i * HIDD, xr, HIDD, HIDD);
    gemm_bf16_wmma<0><<<gemm_grid_e, dim3(256), 0, stream>>>(
        e0_bf, We_bf + (size_t)i * HIDD * HIDD, nullptr, ep, HIDD, HIDD);

    const int ninit = NN * HIDD;
    init_layer_kernel<<<dim3((ninit + 255) / 256), dim3(256), 0, stream>>>(
        agg, amax, denom, b_conv + i * HIDD, ninit, NN * NHEAD);

    attn_logits_kernel<<<dim3(EE), dim3(256), 0, stream>>>(
        xl, xr, ep, src, dst, att + i * HIDD, logits, amax);
    attn_exp_kernel<<<dim3((EE * NHEAD + 255) / 256), dim3(256), 0, stream>>>(
        logits, amax, dst, wbuf, denom, EE * NHEAD);
    aggregate_kernel<<<dim3(EE), dim3(256), 0, stream>>>(xl, wbuf, denom, src, dst, agg);

    ln_kernel<<<dim3(NN), dim3(256), 0, stream>>>(h, agg, ln_g + i * HIDD, ln_b + i * HIDD);
  }

  // Pooling: mean + sum per graph -> gcat[G,512] (bf16, padded)
  pool_init_kernel<<<dim3((GG * HIDD + 255) / 256), dim3(256), 0, stream>>>(
      ssum, cnt, GG * HIDD, GG);
  pool_acc_kernel<<<dim3(NN), dim3(256), 0, stream>>>(h, batch, ssum, cnt);
  pool_fin_kernel<<<dim3(GG), dim3(256), 0, stream>>>(ssum, cnt, gcat);

  // mol_fc: z = silu(g @ W_fc1 + b) @ W_fc2 + b2
  gemm_bf16_wmma<1><<<gemm_grid_g, dim3(256), 0, stream>>>(gcat, Wf1_bf, b_fc1, z1, 512, HIDD);
  cvt(z1, z1_bf, (size_t)GP * HIDD);
  gemm_bf16_wmma<0><<<gemm_grid_g, dim3(256), 0, stream>>>(z1_bf, Wf2_bf, b_fc2, z2, HIDD, HIDD);

  // Task heads
  head1_kernel<<<dim3(GG, NTASK), dim3(128), 0, stream>>>(z2, W_h1, b_h1, h1);
  head2_kernel<<<dim3(GG, NTASK), dim3(64), 0, stream>>>(h1, W_h2, b_h2, h2);
  head3_kernel<<<dim3((GG * NTASK + 255) / 256), dim3(256), 0, stream>>>(
      h2, W_h3, b_h3, out, GG * NTASK);
}
